// GT_5935644803342
// MI455X (gfx1250) — matrix operand order repaired and checked by reading, not compiled, not run on hardware
//
#include <hip/hip_runtime.h>
#include <hip/hip_bf16.h>
#include <stdint.h>

typedef __attribute__((ext_vector_type(16))) __bf16 v16bf;
typedef __attribute__((ext_vector_type(8)))  float  v8f;

#define B_     16
#define N_     448
#define S_     449
#define H_     768
#define HEADS_ 12
#define DK_    64
#define FFN_   3072
#define L_     6
#define BD_    8
#define OUT_   10
#define SP_    480      // padded S for attention (15*32)
#define QT_    29       // ceil(S/16)
#define ROWS_  (B_*S_)  // 7184
#define SCALE_ 0.125f   // DK^-0.5
#define EPS_   1e-5f

// flags for gemm
#define FLG_GELU  1
#define FLG_ENC   2     // encoder row remap: orow = r + r/448

union FragBF { v16bf v; uint32_t u[8]; };

__device__ __forceinline__ int imin(int a, int b) { return a < b ? a : b; }

// pack two f32 -> (bf16(lo) | bf16(hi)<<16), nearest rounding, 3 VALU ops
__device__ __forceinline__ uint32_t f2bf2(float lo, float hi) {
  uint32_t a = __float_as_uint(lo) + 0x8000u;
  uint32_t b = __float_as_uint(hi) + 0x8000u;
  return __builtin_amdgcn_perm(b, a, 0x07060302u); // {b.b3,b.b2,a.b3,a.b2}
}
__device__ __forceinline__ __bf16 f2bf(float f) {
  union { unsigned short s; __bf16 b; } y;
  y.s = (unsigned short)((__float_as_uint(f) + 0x8000u) >> 16);
  return y.b;
}

// CDNA5 async global->LDS copy (ASYNCcnt); vdst = LDS byte offset (addr[31:0])
__device__ __forceinline__ void async_copy_b128(uint32_t lds_off, const void* gptr) {
  asm volatile("global_load_async_to_lds_b128 %0, %1, off"
               :: "v"(lds_off), "v"(gptr) : "memory");
}
__device__ __forceinline__ void wait_async0() {
  asm volatile("s_wait_asynccnt 0" ::: "memory");
}

// ---------------------------------------------------------------------------
// Generic GEMM: C[M,N] = act(A[M,K] @ Wp[K,N] + bias) (+ R residual)
// A: bf16 row-major. Wp: pair-packed bf16 words Wp[(k/2)*N + n] = {k, k+1}.
// Out-of-range tile rows/cols CLAMPED (outputs discarded). K multiple of 32.
// Block: 256 thr = 8 waves; tile 128x64; wave 32x32 = 2x2 WMMA 16x16x32.
// Tiles fetched with global_load_async_to_lds_b128.
// ---------------------------------------------------------------------------
__launch_bounds__(256)
__global__ void gemm_bf16_wmma(const __bf16* __restrict__ A, int a_rs,
                               const uint32_t* __restrict__ Wp,
                               const float* __restrict__ bias,
                               const float* R,
                               float* Cf, __bf16* Cb,
                               int c_rs, int M, int N, int K, int flags)
{
  __shared__ alignas(16) __bf16   As[128][40];   // [row][k], 80B rows
  __shared__ alignas(16) uint32_t BsP[16][72];   // [kpair][col], 288B rows
  const int tid  = threadIdx.x;
  const int lane = tid & 31;
  const int half = lane >> 4, nl = lane & 15;
  const int wave = tid >> 5;
  const int rm0  = (wave >> 1) * 32;
  const int cn0  = (wave & 1) * 32;
  const int m0   = blockIdx.y * 128, n0 = blockIdx.x * 64;
  const bool fullN = (n0 + 64 <= N);

  v8f zero = {};
  v8f acc00 = zero, acc01 = zero, acc10 = zero, acc11 = zero;

  for (int k0 = 0; k0 < K; k0 += 32) {
    // ---- A tile 128x32 bf16: 512 b128 chunks, async copy ----
    #pragma unroll
    for (int i = 0; i < 2; ++i) {
      int slot = tid + i * 256;
      int row = slot >> 2, c8 = (slot & 3) * 8;
      int gr = imin(m0 + row, M - 1);
      async_copy_b128((uint32_t)(uintptr_t)&As[row][c8],
                      A + (size_t)gr * a_rs + k0 + c8);
    }
    // ---- B tile 16 pair-rows x 64: 256 b128 chunks, async copy ----
    if (fullN) {
      int r = tid >> 4, cq = (tid & 15) * 4;
      async_copy_b128((uint32_t)(uintptr_t)&BsP[r][cq],
                      Wp + (size_t)((k0 >> 1) + r) * N + n0 + cq);
    } else {
      #pragma unroll
      for (int i = 0; i < 4; ++i) {
        int e = tid + i * 256;
        int r = e >> 6, c = e & 63;
        int gc = imin(n0 + c, N - 1);
        BsP[r][c] = Wp[(size_t)((k0 >> 1) + r) * N + gc];
      }
    }
    wait_async0();
    __syncthreads();
    if (k0 + 32 < K)
      __builtin_prefetch(&A[(size_t)(m0 + rm0) * a_rs + k0 + 32], 0, 1);

    FragBF a0, a1, b0v, b1v;
    #pragma unroll
    for (int v = 0; v < 8; ++v) {
      int kb = ((v & 4) ? 16 : 0) + half * 8 + (v & 3) * 2;  // A layout (ISA 7.12.2)
      a0.u[v] = *(const uint32_t*)&As[rm0 + nl][kb];
      a1.u[v] = *(const uint32_t*)&As[rm0 + 16 + nl][kb];
      int kp = ((v & 4) ? 8 : 0) + half * 4 + (v & 3);        // B pair row
      b0v.u[v] = BsP[kp][cn0 + nl];
      b1v.u[v] = BsP[kp][cn0 + 16 + nl];
    }
    acc00 = __builtin_amdgcn_wmma_f32_16x16x32_bf16(false, a0.v, false, b0v.v, (short)0, acc00, false, false);
    acc01 = __builtin_amdgcn_wmma_f32_16x16x32_bf16(false, a0.v, false, b1v.v, (short)0, acc01, false, false);
    acc10 = __builtin_amdgcn_wmma_f32_16x16x32_bf16(false, a1.v, false, b0v.v, (short)0, acc10, false, false);
    acc11 = __builtin_amdgcn_wmma_f32_16x16x32_bf16(false, a1.v, false, b1v.v, (short)0, acc11, false, false);
    __syncthreads();
  }

  #pragma unroll
  for (int sm = 0; sm < 2; ++sm) {
    #pragma unroll
    for (int sn = 0; sn < 2; ++sn) {
      v8f acc = sm == 0 ? (sn == 0 ? acc00 : acc01) : (sn == 0 ? acc10 : acc11);
      #pragma unroll
      for (int r = 0; r < 8; ++r) {
        int m = rm0 + sm * 16 + r + half * 8;    // D layout: m = r + 8*half
        int n = cn0 + sn * 16 + nl;
        int grow = m0 + m, gcol = n0 + n;
        if (grow < M && gcol < N) {
          float vv = acc[r];
          if (bias) vv += bias[gcol];
          if (flags & FLG_GELU) vv = 0.5f * vv * (1.0f + erff(vv * 0.70710678118f));
          size_t orow = (flags & FLG_ENC) ? (size_t)grow + (size_t)(grow / N_)
                                          : (size_t)grow;
          size_t ci = orow * (size_t)c_rs + gcol;
          if (R) vv += R[ci];
          if (Cb) Cb[ci] = f2bf(vv); else Cf[ci] = vv;
        }
      }
    }
  }
}

// ---------------------------------------------------------------------------
// Attention: one wave per (b, head, 16-row q tile). scores = QK^T*s + bias,
// softmax over padded 480 cols, O = P @ V. All matmuls via WMMA bf16.
// ---------------------------------------------------------------------------
__launch_bounds__(32)
__global__ void attn_wmma(const __bf16* __restrict__ Q, const __bf16* __restrict__ Kbuf,
                          const __bf16* __restrict__ Vbuf, const float* __restrict__ Bias,
                          __bf16* __restrict__ O)
{
  __shared__ float    sc[16][SP_];
  __shared__ __bf16   pb[16][SP_];
  __shared__ uint32_t VsP[16][68];   // pair-packed V chunk: [kpair][d]
  const int lane = threadIdx.x & 31;
  const int half = lane >> 4, nl = lane & 15;
  const int qt = blockIdx.x, hh = blockIdx.y, bb = blockIdx.z;
  const int q0 = qt * 16;

  #pragma unroll 1
  for (int e = lane; e < 16 * SP_; e += 32) ((float*)sc)[e] = -1.0e30f;

  int qrow = imin(q0 + nl, S_ - 1);
  const __bf16* qp = Q + ((size_t)(bb * S_ + qrow)) * H_ + hh * DK_;
  FragBF aq0, aq1;
  #pragma unroll
  for (int v = 0; v < 8; ++v) {
    int kb = ((v & 4) ? 16 : 0) + half * 8 + (v & 3) * 2;
    aq0.u[v] = *(const uint32_t*)(qp + kb);
    aq1.u[v] = *(const uint32_t*)(qp + 32 + kb);
  }
  __syncthreads();

  #pragma unroll 1
  for (int j = 0; j < QT_; ++j) {                    // score tiles along k
    int k0 = j * 16;
    int kcol = k0 + nl;
    bool cvalid = kcol < S_;
    const __bf16* kp = Kbuf + ((size_t)(bb * S_ + imin(kcol, S_ - 1))) * H_ + hh * DK_;
    FragBF bk0, bk1;
    #pragma unroll
    for (int v = 0; v < 8; ++v) {
      int kr = ((v & 4) ? 16 : 0) + half * 8 + (v & 3) * 2;
      bk0.u[v] = *(const uint32_t*)(kp + kr);
      bk1.u[v] = *(const uint32_t*)(kp + 32 + kr);
    }
    v8f c = {};
    c = __builtin_amdgcn_wmma_f32_16x16x32_bf16(false, aq0.v, false, bk0.v, (short)0, c, false, false);
    c = __builtin_amdgcn_wmma_f32_16x16x32_bf16(false, aq1.v, false, bk1.v, (short)0, c, false, false);
    #pragma unroll
    for (int r = 0; r < 8; ++r) {
      int m = r + half * 8;
      int grow = q0 + m;
      float s = -1.0e30f;
      if (cvalid && grow < S_)
        s = c[r] * SCALE_ + Bias[(((size_t)bb * HEADS_ + hh) * S_ + grow) * (size_t)S_ + kcol];
      sc[m][k0 + nl] = s;
    }
  }
  __syncthreads();

  #pragma unroll 1
  for (int m = 0; m < 16; ++m) {                     // wave-level softmax per row
    float mx = -3.0e38f;
    #pragma unroll 1
    for (int cix = lane; cix < SP_; cix += 32) mx = fmaxf(mx, sc[m][cix]);
    #pragma unroll
    for (int o = 16; o > 0; o >>= 1) mx = fmaxf(mx, __shfl_xor(mx, o, 32));
    float sum = 0.f;
    #pragma unroll 1
    for (int cix = lane; cix < SP_; cix += 32) { float e = __expf(sc[m][cix] - mx); sc[m][cix] = e; sum += e; }
    #pragma unroll
    for (int o = 16; o > 0; o >>= 1) sum += __shfl_xor(sum, o, 32);
    float inv = 1.0f / sum;
    #pragma unroll 1
    for (int cix = lane * 2; cix < SP_; cix += 64)   // packed bf16 pair stores
      *(uint32_t*)&pb[m][cix] = f2bf2(sc[m][cix] * inv, sc[m][cix + 1] * inv);
  }
  __syncthreads();

  v8f oacc[4]; v8f zero = {};
  #pragma unroll
  for (int t = 0; t < 4; ++t) oacc[t] = zero;

  #pragma unroll 1
  for (int kc = 0; kc < SP_ / 32; ++kc) {            // O = P @ V, K = 480
    // stage V rows [kc*32, kc*32+32) pair-packed: VsP[p][d] = {V[2p][d], V[2p+1][d]}
    {
      int r0g = imin(kc * 32 + 2 * nl,     S_ - 1);  // P==0 on pads -> value irrelevant
      int r1g = imin(kc * 32 + 2 * nl + 1, S_ - 1);
      const uint32_t* v0 = (const uint32_t*)(Vbuf + ((size_t)(bb * S_ + r0g)) * H_ + hh * DK_ + half * 32);
      const uint32_t* v1 = (const uint32_t*)(Vbuf + ((size_t)(bb * S_ + r1g)) * H_ + hh * DK_ + half * 32);
      #pragma unroll
      for (int w = 0; w < 16; ++w) {                 // 16 words = 32 d values
        uint32_t w0 = v0[w], w1 = v1[w];
        VsP[nl][half * 32 + 2 * w]     = __builtin_amdgcn_perm(w1, w0, 0x05040100u);
        VsP[nl][half * 32 + 2 * w + 1] = __builtin_amdgcn_perm(w1, w0, 0x07060302u);
      }
    }
    __syncthreads();
    FragBF ap;
    #pragma unroll
    for (int v = 0; v < 8; ++v) {
      int kb = kc * 32 + ((v & 4) ? 16 : 0) + half * 8 + (v & 3) * 2;
      ap.u[v] = *(const uint32_t*)&pb[nl][kb];
    }
    #pragma unroll
    for (int t = 0; t < 4; ++t) {
      FragBF bv;
      #pragma unroll
      for (int v = 0; v < 8; ++v)
        bv.u[v] = VsP[((v & 4) ? 8 : 0) + half * 4 + (v & 3)][t * 16 + nl];
      oacc[t] = __builtin_amdgcn_wmma_f32_16x16x32_bf16(false, ap.v, false, bv.v, (short)0, oacc[t], false, false);
    }
    __syncthreads();
  }
  #pragma unroll
  for (int t = 0; t < 4; ++t) {
    #pragma unroll
    for (int r = 0; r < 8; ++r) {
      int m = r + half * 8;
      int grow = q0 + m;
      if (grow < S_)
        O[((size_t)(bb * S_ + grow)) * H_ + hh * DK_ + t * 16 + nl] = f2bf(oacc[t][r]);
    }
  }
}

// ---------------------------------------------------------------------------
// LayerNorm: fp32 in -> bf16 out (pair-packed stores)
__launch_bounds__(256)
__global__ void ln_kernel(const float* __restrict__ X, const float* __restrict__ g,
                          const float* __restrict__ bta, __bf16* __restrict__ Y, int rows)
{
  int row = blockIdx.x * 8 + (threadIdx.x >> 5);
  int lane = threadIdx.x & 31;
  if (row >= rows) return;
  const float* xr = X + (size_t)row * H_;
  float2 v2[H_ / 64];
  float s = 0.f;
  #pragma unroll
  for (int i = 0; i < H_ / 64; ++i) {
    v2[i] = *(const float2*)&xr[lane * 2 + i * 64];
    s += v2[i].x + v2[i].y;
  }
  #pragma unroll
  for (int o = 16; o > 0; o >>= 1) s += __shfl_xor(s, o, 32);
  float mean = s * (1.0f / H_);
  float s2 = 0.f;
  #pragma unroll
  for (int i = 0; i < H_ / 64; ++i) {
    float dx = v2[i].x - mean, dy = v2[i].y - mean;
    s2 += dx * dx + dy * dy;
  }
  #pragma unroll
  for (int o = 16; o > 0; o >>= 1) s2 += __shfl_xor(s2, o, 32);
  float rstd = rsqrtf(s2 * (1.0f / H_) + EPS_);
  #pragma unroll
  for (int i = 0; i < H_ / 64; ++i) {
    int c = lane * 2 + i * 64;
    float a = (v2[i].x - mean) * rstd * g[c]     + bta[c];
    float b = (v2[i].y - mean) * rstd * g[c + 1] + bta[c + 1];
    *(uint32_t*)&Y[(size_t)row * H_ + c] = f2bf2(a, b);
  }
}

__launch_bounds__(256)
__global__ void bias_kernel(const float* __restrict__ attn_bias, const float* __restrict__ gvd,
                            const float* __restrict__ Wb, const float* __restrict__ bbv,
                            float* __restrict__ Bias)
{
  uint32_t idx = blockIdx.x * 256 + threadIdx.x;
  if (idx >= (uint32_t)B_ * S_ * S_) return;
  int k = (int)(idx % S_);
  uint32_t t = idx / S_;
  int q = (int)(t % S_);
  int b = (int)(t / S_);
  const float* src;
  if (k >= N_)      src = gvd + (size_t)(k - N_) * BD_;
  else if (q >= N_) src = gvd + (size_t)(q - N_) * BD_;
  else              src = attn_bias + (((size_t)b * N_ + q) * N_ + k) * BD_;
  float gb[BD_];
  #pragma unroll
  for (int d = 0; d < BD_; ++d) gb[d] = src[d];
  #pragma unroll
  for (int h2 = 0; h2 < HEADS_; ++h2) {
    float a = bbv[h2];
    #pragma unroll
    for (int d = 0; d < BD_; ++d) a += gb[d] * Wb[d * HEADS_ + h2];
    Bias[(((size_t)b * HEADS_ + h2) * S_ + q) * (size_t)S_ + k] = a;
  }
}

// fp32 [2*K2 x Nw] -> pair-packed bf16 words [K2 x Nw]
__launch_bounds__(256)
__global__ void packw_kernel(const float* __restrict__ W, uint32_t* __restrict__ Wp,
                             uint32_t K2, uint32_t Nw)
{
  uint32_t idx = blockIdx.x * 256 + threadIdx.x;
  if (idx >= K2 * Nw) return;
  uint32_t kp = idx / Nw, n = idx % Nw;
  Wp[idx] = f2bf2(W[(size_t)(2 * kp) * Nw + n], W[(size_t)(2 * kp + 1) * Nw + n]);
}

// fp32 -> bf16 flat (pairs)
__launch_bounds__(256)
__global__ void packx_kernel(const float* __restrict__ X, uint32_t* __restrict__ Xp, uint32_t tot)
{
  uint32_t i = blockIdx.x * 256 + threadIdx.x;
  if (i >= tot) return;
  Xp[i] = f2bf2(X[2 * i], X[2 * i + 1]);
}

__global__ void token_kernel(const float* __restrict__ gt, float* __restrict__ h)
{
  int idx = blockIdx.x * 256 + threadIdx.x;
  if (idx >= B_ * H_) return;
  int b = idx / H_, j = idx % H_;
  h[((size_t)b * S_ + N_) * H_ + j] = gt[j];
}

__global__ void logsoftmax_kernel(const float* __restrict__ lg, float* __restrict__ out)
{
  int b = threadIdx.x;
  if (b >= B_) return;
  float mx = -3.0e38f;
  for (int j = 0; j < OUT_; ++j) mx = fmaxf(mx, lg[b * OUT_ + j]);
  float sum = 0.f;
  for (int j = 0; j < OUT_; ++j) sum += __expf(lg[b * OUT_ + j] - mx);
  float lse = mx + __logf(sum);
  for (int j = 0; j < OUT_; ++j) out[b * OUT_ + j] = lg[b * OUT_ + j] - lse;
}

// ---------------------------------------------------------------------------
extern "C" void kernel_launch(void* const* d_in, const int* in_sizes, int n_in,
                              void* d_out, int out_size, void* d_ws, size_t ws_size,
                              hipStream_t stream)
{
  const float* attn_bias = (const float*)d_in[0];
  const float* x         = (const float*)d_in[1];
  const float* enc_W     = (const float*)d_in[2];
  const float* enc_b     = (const float*)d_in[3];
  const float* graph_tok = (const float*)d_in[4];
  const float* gvd       = (const float*)d_in[5];
  const float* ln1_g     = (const float*)d_in[6];
  const float* ln1_b     = (const float*)d_in[7];
  const float* Wq        = (const float*)d_in[8];
  const float* bq        = (const float*)d_in[9];
  const float* Wk        = (const float*)d_in[10];
  const float* bk        = (const float*)d_in[11];
  const float* Wv        = (const float*)d_in[12];
  const float* bv        = (const float*)d_in[13];
  const float* Wbias     = (const float*)d_in[14];
  const float* bbias     = (const float*)d_in[15];
  const float* Wo        = (const float*)d_in[16];
  const float* bo        = (const float*)d_in[17];
  const float* ln2_g     = (const float*)d_in[18];
  const float* ln2_b     = (const float*)d_in[19];
  const float* W1        = (const float*)d_in[20];
  const float* b1        = (const float*)d_in[21];
  const float* W2        = (const float*)d_in[22];
  const float* b2        = (const float*)d_in[23];
  const float* fln_g     = (const float*)d_in[24];
  const float* fln_b     = (const float*)d_in[25];
  const float* out_W     = (const float*)d_in[26];
  const float* out_b     = (const float*)d_in[27];

  char* wsb = (char*)d_ws;
  size_t off = 0;
  auto alloc = [&](size_t bytes) -> char* {
    char* p = wsb + off; off = (off + bytes + 255) & ~(size_t)255; return p;
  };
  float*    h       = (float*)   alloc((size_t)ROWS_ * H_ * 4);
  __bf16*   ybf     = (__bf16*)  alloc((size_t)ROWS_ * H_ * 2);
  __bf16*   qbuf    = (__bf16*)  alloc((size_t)ROWS_ * H_ * 2);
  __bf16*   kbuf    = (__bf16*)  alloc((size_t)ROWS_ * H_ * 2);
  __bf16*   vbuf    = (__bf16*)  alloc((size_t)ROWS_ * H_ * 2);
  __bf16*   obuf    = (__bf16*)  alloc((size_t)ROWS_ * H_ * 2);
  __bf16*   ffnb    = (__bf16*)  alloc((size_t)ROWS_ * FFN_ * 2);
  float*    biasbuf = (float*)   alloc((size_t)B_ * HEADS_ * S_ * S_ * 4);
  float*    logits  = (float*)   alloc((size_t)B_ * OUT_ * 4);
  uint32_t* xp      = (uint32_t*)alloc((size_t)B_ * N_ * 128 / 2 * 4);
  uint32_t* encp    = (uint32_t*)alloc((size_t)64 * H_ * 4);
  uint32_t* wqp     = (uint32_t*)alloc((size_t)L_ * H_ / 2 * H_ * 4);
  uint32_t* wkp     = (uint32_t*)alloc((size_t)L_ * H_ / 2 * H_ * 4);
  uint32_t* wvp     = (uint32_t*)alloc((size_t)L_ * H_ / 2 * H_ * 4);
  uint32_t* wop     = (uint32_t*)alloc((size_t)L_ * H_ / 2 * H_ * 4);
  uint32_t* w1p     = (uint32_t*)alloc((size_t)L_ * H_ / 2 * FFN_ * 4);
  uint32_t* w2p     = (uint32_t*)alloc((size_t)L_ * FFN_ / 2 * H_ * 4);
  uint32_t* outp    = (uint32_t*)alloc((size_t)(H_ / 2) * OUT_ * 4);
  (void)in_sizes; (void)n_in; (void)out_size; (void)ws_size;

  dim3 blk(256);
  auto nb = [](uint32_t tot) { return (tot + 255u) / 256u; };

  // one-time (per launch) bf16 pre-packing of inputs/weights
  packx_kernel<<<nb((uint32_t)B_ * N_ * 64), blk, 0, stream>>>(x, xp, (uint32_t)B_ * N_ * 64);
  packw_kernel<<<nb(64u * H_), blk, 0, stream>>>(enc_W, encp, 64, H_);
  packw_kernel<<<nb((uint32_t)L_ * H_ / 2 * H_), blk, 0, stream>>>(Wq, wqp, L_ * H_ / 2, H_);
  packw_kernel<<<nb((uint32_t)L_ * H_ / 2 * H_), blk, 0, stream>>>(Wk, wkp, L_ * H_ / 2, H_);
  packw_kernel<<<nb((uint32_t)L_ * H_ / 2 * H_), blk, 0, stream>>>(Wv, wvp, L_ * H_ / 2, H_);
  packw_kernel<<<nb((uint32_t)L_ * H_ / 2 * H_), blk, 0, stream>>>(Wo, wop, L_ * H_ / 2, H_);
  packw_kernel<<<nb((uint32_t)L_ * H_ / 2 * FFN_), blk, 0, stream>>>(W1, w1p, L_ * H_ / 2, FFN_);
  packw_kernel<<<nb((uint32_t)L_ * FFN_ / 2 * H_), blk, 0, stream>>>(W2, w2p, L_ * FFN_ / 2, H_);
  packw_kernel<<<nb((uint32_t)(H_ / 2) * OUT_), blk, 0, stream>>>(out_W, outp, H_ / 2, OUT_);

  // Encoder: [B*N,128] @ [128,768] -> h rows remapped b*S+s
  {
    dim3 g((H_ + 63) / 64, (B_ * N_ + 127) / 128);
    gemm_bf16_wmma<<<g, blk, 0, stream>>>((const __bf16*)xp, 128, encp, enc_b, nullptr,
                                          h, nullptr, H_, B_ * N_, H_, 128, FLG_ENC);
  }
  token_kernel<<<(B_ * H_ + 255) / 256, blk, 0, stream>>>(graph_tok, h);

  dim3 gproj((H_ + 63) / 64, (ROWS_ + 127) / 128);
  dim3 gf1((FFN_ + 63) / 64, (ROWS_ + 127) / 128);
  dim3 gattn(QT_, HEADS_, B_);

  for (int l = 0; l < L_; ++l) {
    bias_kernel<<<nb((uint32_t)B_ * S_ * S_), blk, 0, stream>>>(
        attn_bias, gvd, Wbias + (size_t)l * BD_ * HEADS_, bbias + (size_t)l * HEADS_, biasbuf);
    ln_kernel<<<(ROWS_ + 7) / 8, blk, 0, stream>>>(h, ln1_g + l * H_, ln1_b + l * H_, ybf, ROWS_);
    gemm_bf16_wmma<<<gproj, blk, 0, stream>>>(ybf, H_, wqp + (size_t)l * (H_ / 2) * H_, bq + l * H_,
        nullptr, nullptr, qbuf, H_, ROWS_, H_, H_, 0);
    gemm_bf16_wmma<<<gproj, blk, 0, stream>>>(ybf, H_, wkp + (size_t)l * (H_ / 2) * H_, bk + l * H_,
        nullptr, nullptr, kbuf, H_, ROWS_, H_, H_, 0);
    gemm_bf16_wmma<<<gproj, blk, 0, stream>>>(ybf, H_, wvp + (size_t)l * (H_ / 2) * H_, bv + l * H_,
        nullptr, nullptr, vbuf, H_, ROWS_, H_, H_, 0);
    attn_wmma<<<gattn, 32, 0, stream>>>(qbuf, kbuf, vbuf, biasbuf, obuf);
    gemm_bf16_wmma<<<gproj, blk, 0, stream>>>(obuf, H_, wop + (size_t)l * (H_ / 2) * H_, bo + l * H_,
        h, h, nullptr, H_, ROWS_, H_, H_, 0);             // residual into h
    ln_kernel<<<(ROWS_ + 7) / 8, blk, 0, stream>>>(h, ln2_g + l * H_, ln2_b + l * H_, ybf, ROWS_);
    gemm_bf16_wmma<<<gf1, blk, 0, stream>>>(ybf, H_, w1p + (size_t)l * (H_ / 2) * FFN_, b1 + l * FFN_,
        nullptr, nullptr, ffnb, FFN_, ROWS_, FFN_, H_, FLG_GELU);
    gemm_bf16_wmma<<<gproj, blk, 0, stream>>>(ffnb, FFN_, w2p + (size_t)l * (FFN_ / 2) * H_, b2 + l * H_,
        h, h, nullptr, H_, ROWS_, H_, FFN_, 0);           // residual into h
  }
  ln_kernel<<<(ROWS_ + 7) / 8, blk, 0, stream>>>(h, fln_g, fln_b, ybf, ROWS_);
  {
    dim3 g(1, 1);  // readout: rows b*S+0 via a_rs = S*H
    gemm_bf16_wmma<<<g, blk, 0, stream>>>(ybf, S_ * H_, outp, out_b, nullptr,
                                          logits, nullptr, OUT_, B_, OUT_, H_, 0);
  }
  logsoftmax_kernel<<<1, 32, 0, stream>>>(logits, (float*)d_out);
}
